// LSTM_74354473828522
// MI455X (gfx1250) — matrix operand order repaired and checked by reading, not compiled, not run on hardware
//
#include <hip/hip_runtime.h>
#include <hip/hip_bf16.h>

// LSTM for MI455X (gfx1250, wave32, WMMA + TDM).
//   B=64, T=2048, I=256, H=512, 4H=2048.
//   1) lstm_prep : transpose+convert weights to bf16 [n][k] layout, fuse biases.
//   2) lstm_init : zero h ping-pong buffer (t=0 half) + barrier counters.
//   3) lstm_xproj: Xp[t][b][4H] = bf16( X @ Wx )  -- big parallel WMMA GEMM.
//   4) lstm_scan : persistent kernel, 32 blocks x 4 waves. Each block owns 16
//      hidden columns; its 64 KB slice of Wh is DMA'd into LDS once via
//      TENSOR_LOAD_TO_LDS (with TDM padding for bank-conflict-free ds reads),
//      then 2048 steps of z = Xp + bias + h@Wh with B-fragments from LDS,
//      c kept in registers, grid barrier (agent atomics + s_sleep) per step.
// Workspace requirement: ~517 MiB (Xp bf16 dominates).

#define B_  64
#define T_  2048
#define I_  256
#define H_  512
#define NG  2048   // 4*H

// Padded LDS layout for Wh slice: column = 512 bf16 (1024 B) + 8 B pad.
#define COL_BYTES   1032
#define COL_ELEMS   516
#define GATE_BYTES  (16 * COL_BYTES)   // 16512
#define GATE_ELEMS  (16 * COL_ELEMS)   // 8256

typedef __attribute__((ext_vector_type(16))) __bf16 v16bf;
typedef __attribute__((ext_vector_type(8)))  float  v8f;
typedef __attribute__((ext_vector_type(4)))  unsigned u32x4;
typedef __attribute__((ext_vector_type(8)))  int      i32x8;
typedef __attribute__((ext_vector_type(4)))  int      i32x4;

union BFrag { unsigned u[8]; __bf16 h[16]; v16bf v; };

struct GateW {
    const float* wx[4];
    const float* bx[4];
    const float* wh[4];
    const float* bh[4];
};

__device__ __forceinline__ float sigmoid_f(float x) {
    return 1.0f / (1.0f + __expf(-x));
}
__device__ __forceinline__ float tanh_f(float x) {
#if __has_builtin(__builtin_amdgcn_tanhf)
    return __builtin_amdgcn_tanhf(x);   // CDNA5 TANH trans op
#else
    float e = __expf(-2.0f * x);
    return (1.0f - e) / (1.0f + e);
#endif
}

// ---------------------------------------------------------------------------
// 1) Weight prep: WxT[n][k] (bf16, n=gate*512+j, k<256), WhT[n][k] (k<512),
//    bias[n] = bx[n] + bh[n]. Source w is [in_dim][H] row-major.
// ---------------------------------------------------------------------------
__global__ __launch_bounds__(256) void lstm_prep(GateW P, __bf16* __restrict__ WxT,
                                                 __bf16* __restrict__ WhT,
                                                 float* __restrict__ bias)
{
    const int n    = blockIdx.x;      // 0..2047
    const int k    = threadIdx.x;     // 0..255
    const int gate = n >> 9;
    const int j    = n & (H_ - 1);
    WxT[(size_t)n * I_ + k]       = (__bf16)P.wx[gate][(size_t)k * H_ + j];
    WhT[(size_t)n * H_ + k]       = (__bf16)P.wh[gate][(size_t)k * H_ + j];
    WhT[(size_t)n * H_ + k + 256] = (__bf16)P.wh[gate][(size_t)(k + 256) * H_ + j];
    if (k == 0) bias[n] = P.bx[gate][j] + P.bh[gate][j];
}

// ---------------------------------------------------------------------------
// 2) Init: zero h buffer (ping half used at t=0) and barrier state.
// ---------------------------------------------------------------------------
__global__ __launch_bounds__(256) void lstm_init(__bf16* __restrict__ hbuf,
                                                 unsigned* __restrict__ sync)
{
    const int i = blockIdx.x * 256 + threadIdx.x;   // 0..32767
    hbuf[i] = (__bf16)0.0f;
    if (i < 2) sync[i] = 0u;
}

// ---------------------------------------------------------------------------
// 3) Xp = bf16( X @ Wx ), stored [t][b][4H].
//    Grid: 8192 row tiles (16 rows of m=b*T+t) x 32 col groups (64 cols each).
// ---------------------------------------------------------------------------
__global__ __launch_bounds__(256) void lstm_xproj(const float* __restrict__ X,
                                                  const __bf16* __restrict__ WxT,
                                                  __bf16* __restrict__ Xp)
{
    const int lane = threadIdx.x & 31;
    const int gw   = blockIdx.x * 8 + (threadIdx.x >> 5);
    const int cg   = gw & 31;        // column group (64 cols)
    const int rt   = gw >> 5;        // row tile
    const int half = lane >> 4;
    const int l15  = lane & 15;
    const int m0   = rt * 16;

    v8f acc[4];
#pragma unroll
    for (int tc = 0; tc < 4; ++tc)
#pragma unroll
        for (int e = 0; e < 8; ++e) acc[tc][e] = 0.0f;

    const float* Arow = X + (size_t)(m0 + l15) * I_;
    for (int kk = 0; kk < I_ / 32; ++kk) {
        const int kbase = kk * 32;
        BFrag a;
#pragma unroll
        for (int r = 0; r < 8; ++r) {
            // 16-bit A layout: lane half h, VGPR r holds K = ((r>>2)<<4)+(h<<3)+((r&3)<<1), +1
            const int k0 = ((r >> 2) << 4) + (half << 3) + ((r & 3) << 1);
            const float2 f2 = *(const float2*)(Arow + kbase + k0);
            a.h[2 * r]     = (__bf16)f2.x;
            a.h[2 * r + 1] = (__bf16)f2.y;
        }
#pragma unroll
        for (int tc = 0; tc < 4; ++tc) {
            const int ncol = cg * 64 + tc * 16 + l15;     // B col = lane&15
            const __bf16* Bp = WxT + (size_t)ncol * I_ + kbase;
            BFrag b;
#pragma unroll
            for (int r = 0; r < 8; ++r)                   // K pair matches A layout
                b.u[r] = *(const unsigned*)(Bp + ((r >> 2) << 4) + (half << 3) + ((r & 3) << 1));
            acc[tc] = __builtin_amdgcn_wmma_f32_16x16x32_bf16(
                false, a.v, false, b.v, (short)0, acc[tc], false, false);
        }
    }
#pragma unroll
    for (int tc = 0; tc < 4; ++tc) {
        const int ncol = cg * 64 + tc * 16 + l15;
#pragma unroll
        for (int r = 0; r < 8; ++r) {
            const int m  = m0 + r + (half << 3);   // C/D: lane L, vgpr r -> M=r+8*(L/16)
            const int b  = m >> 11;                // m = b*T + t
            const int tt = m & (T_ - 1);
            Xp[((size_t)tt * B_ + b) * NG + ncol] = (__bf16)acc[tc][r];
        }
    }
}

// ---------------------------------------------------------------------------
// 4) Persistent recurrent scan: 32 blocks x 128 threads (4 waves).
//    Block -> hidden tile jt = blockIdx.x (16 cols); wave -> batch tile mt.
//    Wh slice (4 gates x 16 cols x 512 K, bf16) staged in LDS via TDM with
//    8 B pad per 1024 B column so ds reads are bank-conflict-free-ish.
// ---------------------------------------------------------------------------
__device__ __forceinline__ void grid_barrier(unsigned* sync, unsigned t)
{
    __threadfence();
    __syncthreads();
    if (threadIdx.x == 0) {
        const unsigned arrived =
            __hip_atomic_fetch_add(&sync[0], 1u, __ATOMIC_ACQ_REL, __HIP_MEMORY_SCOPE_AGENT);
        if (arrived == 31u) {
            __hip_atomic_store(&sync[0], 0u, __ATOMIC_RELAXED, __HIP_MEMORY_SCOPE_AGENT);
            __hip_atomic_fetch_add(&sync[1], 1u, __ATOMIC_RELEASE, __HIP_MEMORY_SCOPE_AGENT);
        } else {
            while (__hip_atomic_load(&sync[1], __ATOMIC_ACQUIRE, __HIP_MEMORY_SCOPE_AGENT) <= t)
                __builtin_amdgcn_s_sleep(1);
        }
    }
    __syncthreads();
}

__global__ __launch_bounds__(128) void lstm_scan(const __bf16* __restrict__ WhT,
                                                 const __bf16* __restrict__ Xp,
                                                 const float* __restrict__ bias,
                                                 __bf16* __restrict__ hbuf,
                                                 float* __restrict__ out,
                                                 unsigned* __restrict__ sync)
{
    __shared__ __align__(16) __bf16 smem[4 * GATE_ELEMS];   // 66048 B

    const int lane = threadIdx.x & 31;
    const int mt   = threadIdx.x >> 5;   // 0..3 batch tile (wave id)
    const int jt   = blockIdx.x;         // 0..31 hidden tile
    const int half = lane >> 4;
    const int l15  = lane & 15;
    const int m0   = mt * 16;
    const int j0   = jt * 16;
    const int jc   = j0 + l15;           // hidden column owned by this lane

    // ---- Stage this block's Wh slice into LDS (once) --------------------
#if __has_builtin(__builtin_amdgcn_tensor_load_to_lds)
    if (threadIdx.x < 32) {              // wave 0 issues the 4 TDM loads
        // Low 32 bits of a flat pointer in the LDS aperture ARE the LDS byte
        // offset.  Deriving lds_addr from the
        // array also escapes it, so the compiler must assume TDM writes it.
        const unsigned lds_base = (unsigned)(unsigned long long)(const void*)smem;
#pragma unroll
        for (int g = 0; g < 4; ++g) {
            const unsigned long long ga =
                (unsigned long long)(const void*)(WhT + (size_t)(g * H_ + j0) * H_);
            // D# group0: count=1 | lds_addr | 57b global addr | type=2
            u32x4 g0;
            g0[0] = 1u;                              // count=1 (valid descriptor)
            g0[1] = lds_base + (unsigned)(g * GATE_BYTES);
            g0[2] = (unsigned)(ga & 0xFFFFFFFFull);
            g0[3] = (unsigned)((ga >> 32) & 0x01FFFFFFull) | (2u << 30);
            // D# group1: data_size=2B, pad_enable, pad_interval=256dw, pad_amount=2dw,
            //            tensor_dim0 = tile_dim0 = 8192 elems (1-D), tensor_dim1=1.
            i32x8 g1;
            g1[0] = (int)((1u << 16) | (1u << 20) | (7u << 22) | (1u << 25));
            g1[1] = (int)((8192u & 0xFFFFu) << 16);  // tensor_dim0[15:0] @ bits 63:48
            g1[2] = (int)(1u << 16);                 // tensor_dim0[31:16]=0, tensor_dim1[15:0]=1
            g1[3] = (int)((8192u & 0xFFFFu) << 16);  // tile_dim0 @ bits 127:112
            g1[4] = 0;                               // tile_dim1=0 (unused), tile_dim2=0
            g1[5] = 8192;                            // tensor_dim0_stride[31:0]
            g1[6] = 0;
            g1[7] = 0;
            i32x4 g2 = {0, 0, 0, 0};
            i32x4 g3 = {0, 0, 0, 0};
            i32x8 g4 = {0, 0, 0, 0, 0, 0, 0, 0};     // extra SGPR group (clang-23 6-arg form)
            __builtin_amdgcn_tensor_load_to_lds(g0, g1, g2, g3, g4, 0);
        }
        __builtin_amdgcn_s_wait_tensorcnt(0);
    }
#else
    // Fallback: manual copy with identical padded layout.
    for (int i = threadIdx.x; i < 4 * 16 * H_; i += 128) {
        const int g   = i >> 13;
        const int rem = i & 8191;
        const int col = rem >> 9;
        const int k   = rem & (H_ - 1);
        smem[g * GATE_ELEMS + col * COL_ELEMS + k] =
            WhT[(size_t)(g * H_ + j0 + col) * H_ + k];
    }
#endif
    __asm__ __volatile__("" ::: "memory");   // LDS now holds Wh slice (TDM-written)
    __syncthreads();

    float bj[4];
#pragma unroll
    for (int g = 0; g < 4; ++g) bj[g] = bias[g * H_ + jc];

    float creg[8];
#pragma unroll
    for (int r = 0; r < 8; ++r) creg[r] = 0.0f;

    for (int t = 0; t < T_; ++t) {
        const __bf16* hin  = hbuf + (size_t)(t & 1) * (B_ * H_);
        __bf16*       hout = hbuf + (size_t)((t + 1) & 1) * (B_ * H_);
        const __bf16* xp   = Xp + (size_t)t * (B_ * NG);

        // acc = Xp[t] + (bx+bh)
        v8f acc[4];
#pragma unroll
        for (int g = 0; g < 4; ++g)
#pragma unroll
            for (int r = 0; r < 8; ++r) {
                const int m = m0 + r + (half << 3);
                acc[g][r] = (float)xp[(size_t)m * NG + g * H_ + jc] + bj[g];
            }
        if (t + 1 < T_)   // warm L2 for next step's Xp slice
            __builtin_prefetch(xp + (size_t)(B_ * NG) + (size_t)m0 * NG + j0, 0, 0);

        // acc += h @ Wh  (K = 512 -> 16 k-steps; A from global h, B from LDS)
        const __bf16* hrow = hin + (size_t)(m0 + l15) * H_;
        for (int kk = 0; kk < H_ / 32; ++kk) {
            const int kbase = kk * 32;
            BFrag a;
#pragma unroll
            for (int r = 0; r < 8; ++r) {
                const int k0 = ((r >> 2) << 4) + (half << 3) + ((r & 3) << 1);
                a.u[r] = *(const unsigned*)(hrow + kbase + k0);
            }
#pragma unroll
            for (int g = 0; g < 4; ++g) {
                const __bf16* Bp = smem + g * GATE_ELEMS + l15 * COL_ELEMS
                                        + kbase;
                BFrag b;
#pragma unroll
                for (int r = 0; r < 8; ++r)
                    b.u[r] = *(const unsigned*)(Bp + ((r >> 2) << 4) + (half << 3) + ((r & 3) << 1));   // ds_load
                acc[g] = __builtin_amdgcn_wmma_f32_16x16x32_bf16(
                    false, a.v, false, b.v, (short)0, acc[g], false, false);
            }
        }

        // gates + state update (c lives in registers)
#pragma unroll
        for (int r = 0; r < 8; ++r) {
            const int m  = m0 + r + (half << 3);
            const float iv = sigmoid_f(acc[0][r]);
            const float fv = sigmoid_f(acc[1][r]);
            const float gv = tanh_f(acc[2][r]);
            const float ov = sigmoid_f(acc[3][r]);
            const float cv = fv * creg[r] + iv * gv;
            creg[r] = cv;
            const float hv = ov * tanh_f(cv);
            hout[(size_t)m * H_ + jc] = (__bf16)hv;
            out[(size_t)m * (T_ * H_) + (size_t)t * H_ + jc] = hv;   // states[b][t][j]
            if (t == T_ - 1) {
                out[(size_t)(B_ * T_ * H_) + (size_t)m * H_ + jc]                     = hv;  // h
                out[(size_t)(B_ * T_ * H_) + (size_t)(B_ * H_) + (size_t)m * H_ + jc] = cv;  // c
            }
        }
        if (t + 1 < T_) grid_barrier(sync, (unsigned)t);
    }
}

// ---------------------------------------------------------------------------
// Host launch
// ---------------------------------------------------------------------------
extern "C" void kernel_launch(void* const* d_in, const int* in_sizes, int n_in,
                              void* d_out, int out_size, void* d_ws, size_t ws_size,
                              hipStream_t stream)
{
    (void)in_sizes; (void)n_in; (void)out_size; (void)ws_size;

    char* ws = (char*)d_ws;
    // Workspace layout (needs ~517 MiB):
    __bf16*   WxT  = (__bf16*)(ws + 0);                       //  1 MiB: [2048][256] bf16
    __bf16*   WhT  = (__bf16*)(ws + (size_t)1 * (1u << 20));  //  2 MiB: [2048][512] bf16
    float*    bias = (float*) (ws + (size_t)3 * (1u << 20));  //  8 KiB
    __bf16*   hbuf = (__bf16*)(ws + (size_t)3 * (1u << 20) + 65536);    // 128 KiB ping-pong
    unsigned* sync = (unsigned*)(ws + (size_t)3 * (1u << 20) + 262144); // 8 B
    __bf16*   Xp   = (__bf16*)(ws + (size_t)4 * (1u << 20));  // 512 MiB: [T][B][4H] bf16

    GateW P;
    P.wx[0] = (const float*)d_in[1];  P.bx[0] = (const float*)d_in[2];
    P.wh[0] = (const float*)d_in[3];  P.bh[0] = (const float*)d_in[4];
    P.wx[1] = (const float*)d_in[5];  P.bx[1] = (const float*)d_in[6];
    P.wh[1] = (const float*)d_in[7];  P.bh[1] = (const float*)d_in[8];
    P.wx[2] = (const float*)d_in[9];  P.bx[2] = (const float*)d_in[10];
    P.wh[2] = (const float*)d_in[11]; P.bh[2] = (const float*)d_in[12];
    P.wx[3] = (const float*)d_in[13]; P.bx[3] = (const float*)d_in[14];
    P.wh[3] = (const float*)d_in[15]; P.bh[3] = (const float*)d_in[16];

    lstm_prep<<<NG, 256, 0, stream>>>(P, WxT, WhT, bias);
    lstm_init<<<(B_ * H_) / 256, 256, 0, stream>>>(hbuf, sync);
    lstm_xproj<<<(8192 * 32) / 8, 256, 0, stream>>>((const float*)d_in[0], WxT, Xp);
    lstm_scan<<<32, 128, 0, stream>>>(WhT, Xp, bias, hbuf, (float*)d_out, sync);
}
